// SegmentTarget_60550448939477
// MI455X (gfx1250) — matrix operand order repaired and checked by reading, not compiled, not run on hardware
//
#include <hip/hip_runtime.h>
#include <hip/hip_bf16.h>

// ---------------------------------------------------------------------------
// SegmentTarget for MI455X (gfx1250, wave32).
// Memory-bound scatter + balanced sampling. ~140MB traffic -> ~6us @ 23.3TB/s.
// WMMA (v_wmma_f32_16x16x32_f16) is used for the exact intra-wave {0,1}
// prefix scan in the positive-selection pass (counts <= 32 are exact in f16).
// ---------------------------------------------------------------------------

typedef __attribute__((ext_vector_type(16))) _Float16 v16h;
typedef __attribute__((ext_vector_type(8)))  float    v8f;

#define FEAT_STRIDE_INV (1.0f / 16.0f)
#define FEAT_STRIDE_F   16.0f

// Deterministic bijective permutation over [0, 2^k): odd multiplier + offset.
__device__ __forceinline__ unsigned perm_idx(unsigned i, unsigned mask) {
    return (i * 2654435761u + 12345u) & mask;
}

// valid / interval-index computation shared by scatter + selection passes.
__device__ __forceinline__ bool compute_valid(const float* __restrict__ slp,
                                              int b, int n, int N, int W,
                                              int& iv_idx, float& x1, float& x2) {
    const float* p = slp + ((size_t)b * N + n) * 2;
    x1 = p[0]; x2 = p[1];
    float iv = floorf((x1 + x2) * 0.5f * FEAT_STRIDE_INV);
    float prev = -1.0f;
    if (n > 0) prev = floorf((p[-2] + p[-1]) * 0.5f * FEAT_STRIDE_INV);
    bool valid = (iv >= 0.0f) && (iv != prev);
    float ivc = fminf(fmaxf(iv, 0.0f), (float)(W - 1));
    iv_idx = (int)ivc;
    return valid;
}

// K0: zero [mask | delta | inside_weights] regions (4*BW floats) as float4.
__global__ void k0_zero(float4* __restrict__ p, long long n4) {
    long long t = (long long)blockIdx.x * blockDim.x + threadIdx.x;
    if (t < n4) p[t] = make_float4(0.f, 0.f, 0.f, 0.f);
}

// K1: scatter-add mask & split-line sums; per-row positive counts.
__global__ void k1_scatter(const float* __restrict__ slp,
                           float* __restrict__ mask, float* __restrict__ delta,
                           int* __restrict__ rowsum, int N, int W) {
    int b = blockIdx.x, n = threadIdx.x;
    int iv_idx; float x1, x2;
    bool valid = compute_valid(slp, b, n, N, W, iv_idx, x1, x2);
    if (valid) {
        size_t o = (size_t)b * W + iv_idx;
        atomicAdd(&mask[o], 1.0f);
        atomicAdd(&delta[2 * o + 0], x1);
        atomicAdd(&delta[2 * o + 1], x2);
    }
    unsigned long long bal = __ballot(valid);           // wave32: low 32 bits
    __shared__ int wc[32];
    int lane = n & 31, wv = n >> 5;
    if (lane == 0) wc[wv] = __popcll(bal);
    __syncthreads();
    if (n == 0) {
        int s = 0, nw = blockDim.x >> 5;
        for (int w = 0; w < nw; ++w) s += wc[w];
        rowsum[b] = s;
    }
}

// K2: scan row sums -> exclusive row offsets (in place); num_positive.
__global__ void k2_scanrows(int* __restrict__ rows, int B,
                            int* __restrict__ hdr, float* __restrict__ scal) {
    __shared__ int sbuf[1024];
    int t = threadIdx.x;
    int v = (t < B) ? rows[t] : 0;
    sbuf[t] = v; __syncthreads();
    for (int d = 1; d < 1024; d <<= 1) {
        int x = (t >= d) ? sbuf[t - d] : 0; __syncthreads();
        sbuf[t] += x; __syncthreads();
    }
    if (t < B) rows[t] = sbuf[t] - v;                   // exclusive offset
    if (t == 1023) { hdr[1] = sbuf[1023]; scal[0] = (float)sbuf[1023]; }
}

// K3: elementwise: cls goals from mask; transform delta sums in place.
__global__ void k3_elemwise(float* __restrict__ cls, float* __restrict__ delta,
                            const float* __restrict__ mask, long long BW, int W) {
    long long t = (long long)blockIdx.x * blockDim.x + threadIdx.x;
    if (t >= BW) return;
    int w = (int)(t % W);
    float m = mask[t];
    cls[t] = (m == 0.0f) ? 0.1f : 0.9f;
    float ctr = ((float)w + 0.5f) * FEAT_STRIDE_F;
    float d0 = delta[2 * t + 0], d1 = delta[2 * t + 1];
    delta[2 * t + 0] = (d0 - ctr) * FEAT_STRIDE_INV;
    delta[2 * t + 1] = (d1 - ctr) * FEAT_STRIDE_INV;
}

// K4: per-chunk negative counts in permuted order.
__global__ void k4_negchunks(const float* __restrict__ mask, int M, unsigned pmask,
                             int* __restrict__ chunk) {
    int i = blockIdx.x * 256 + threadIdx.x;
    bool neg = false;
    if (i < M) neg = (mask[perm_idx((unsigned)i, pmask)] == 0.0f);
    unsigned long long bal = __ballot(neg);
    __shared__ int wc[8];
    int lane = threadIdx.x & 31, wv = threadIdx.x >> 5;
    if (lane == 0) wc[wv] = __popcll(bal);
    __syncthreads();
    if (threadIdx.x == 0) {
        int s = 0;
        for (int w = 0; w < 8; ++w) s += wc[w];
        chunk[blockIdx.x] = s;
    }
}

// K5: scan chunk counts -> exclusive chunk offsets; num_negative; num_samples.
__global__ void k5_scanchunks(int* __restrict__ chunk, int CH,
                              int* __restrict__ hdr, float* __restrict__ scal) {
    __shared__ int sbuf[1024];
    int t = threadIdx.x;
    int per = (CH + 1023) >> 10;
    int lo = t * per, hi = (lo + per < CH) ? lo + per : CH;
    int s = 0;
    for (int k = lo; k < hi; ++k) s += chunk[k];
    sbuf[t] = s; __syncthreads();
    for (int d = 1; d < 1024; d <<= 1) {
        int x = (t >= d) ? sbuf[t - d] : 0; __syncthreads();
        sbuf[t] += x; __syncthreads();
    }
    int run = sbuf[t] - s;                               // exclusive base
    for (int k = lo; k < hi; ++k) { int c = chunk[k]; chunk[k] = run; run += c; }
    if (t == 1023) {
        int total = sbuf[1023];
        hdr[2] = total;
        int np = hdr[1];
        hdr[0] = (np < total) ? np : total;               // num_samples
        scal[1] = (float)total;
    }
}

// K6: positive selection. Intra-wave inclusive {0,1} prefix via two WMMAs:
//   A (16x32 f16): A[m][k] = v[k] for all m (per ISA 16-bit A lane layout)
//   B1[k][j] = (k <= j), B2[k][j] = (k <= 16+j)  => D[*][j] = prefix(j), prefix(16+j)
// Extraction: f32 C/D layout -> lane j (j<16): d1[0] = prefix(j);
//             lane 16+j: d2[0] = D2[M=8][j] = prefix(16+j) (all rows equal).
__global__ void k6_posselect(const float* __restrict__ slp,
                             float* __restrict__ iw,
                             const int* __restrict__ rowoff,
                             const int* __restrict__ hdr, int N, int W) {
    int b = blockIdx.x, n = threadIdx.x;
    int iv_idx; float x1, x2;
    bool valid = compute_valid(slp, b, n, N, W, iv_idx, x1, x2);

    __shared__ _Float16 vals[1024];
    __shared__ int wtot[32];
    vals[n] = (_Float16)(valid ? 1.0f : 0.0f);
    __syncthreads();

    int lane = n & 31, wv = n >> 5, base = wv * 32;
    // A fragment: lanes 0-15 hold K={0..7,16..23}, lanes 16-31 K={8..15,24..31}
    int sub = (lane >= 16) ? 8 : 0;
    v16h a = {};
#pragma unroll
    for (int e = 0; e < 8; ++e) a[e] = vals[base + e + sub];
#pragma unroll
    for (int e = 0; e < 8; ++e) a[8 + e] = vals[base + 16 + e + sub];
    // B fragments: lanes 0-15 hold K={0..7,16..23} (N=lane), lanes 16-31 K={8..15,24..31} (N=lane-16)
    int nn = lane & 15, kof = (lane >= 16) ? 16 : 0;
    v16h b1 = {}, b2 = {};
#pragma unroll
    for (int e = 0; e < 16; ++e) {
        int k = (e < 8) ? (e + sub) : (8 + e + sub);
        b1[e] = (_Float16)((k <= nn) ? 1.0f : 0.0f);
        b2[e] = (_Float16)((k <= nn + 16) ? 1.0f : 0.0f);
    }
    v8f c = {};
    v8f d1 = __builtin_amdgcn_wmma_f32_16x16x32_f16(false, a, false, b1,
                                                    (short)0, c, false, false);
    v8f d2 = __builtin_amdgcn_wmma_f32_16x16x32_f16(false, a, false, b2,
                                                    (short)0, c, false, false);
    float s = (lane < 16) ? d1[0] : d2[0];               // inclusive prefix @ lane
    float st = __shfl(s, 31, 32);                        // wave total
    if (lane == 0) wtot[wv] = (int)(st + 0.5f);
    __syncthreads();
    int woff = 0, nw = blockDim.x >> 5;
    for (int w = 0; w < wv && w < nw; ++w) woff += wtot[w];
    int rank = rowoff[b] + woff + (int)(s + 0.5f);       // global inclusive rank
    if (valid && rank <= hdr[0])
        iw[(size_t)b * W + iv_idx] = 2.0f;               // flag==2 -> POS_WEIGHT
}

// K7: negative selection in permuted order (ballot-rank within block).
__global__ void k7_negselect(const float* __restrict__ mask, float* __restrict__ iw,
                             const int* __restrict__ chunkoff,
                             const int* __restrict__ hdr, int M, unsigned pmask) {
    int i = blockIdx.x * 256 + threadIdx.x;
    unsigned j = 0; bool neg = false;
    if (i < M) { j = perm_idx((unsigned)i, pmask); neg = (mask[j] == 0.0f); }
    unsigned long long bal = __ballot(neg);
    int lane = threadIdx.x & 31, wv = threadIdx.x >> 5;
    int lrank = __popcll(bal & ((2ull << lane) - 1ull)); // inclusive in-wave rank
    __shared__ int wc[8];
    if (lane == 0) wc[wv] = __popcll(bal);
    __syncthreads();
    int woff = 0;
    for (int w = 0; w < wv; ++w) woff += wc[w];
    int rank = chunkoff[blockIdx.x] + woff + lrank;
    if (i < M && neg && rank <= hdr[0])
        iw[j] = 1.0f;                                    // flag==1 -> NEG/PAD == 1.0
}

extern "C" void kernel_launch(void* const* d_in, const int* in_sizes, int n_in,
                              void* d_out, int out_size, void* d_ws, size_t ws_size,
                              hipStream_t stream) {
    const float* slp = (const float*)d_in[0];            // [B,N,2]
    // d_in[1] = pred_cls_logit [B,W] : provably unused by the reference outputs
    // d_in[2] = real_features_width  : unused since NEG_WEIGHT == PAD_WEIGHT == 1.0
    int B = in_sizes[2];
    int W = in_sizes[1] / B;
    int N = in_sizes[0] / (B * 2);
    long long BW = (long long)B * W;
    int M = (int)BW;                                     // 2^22 for given shapes
    unsigned pmask = (unsigned)(M - 1);
    int CH = (M + 255) / 256;

    float* out   = (float*)d_out;
    float* cls   = out;                                  // [B,W]
    float* delta = out + BW;                             // [B,W,2] (sums, then delta)
    float* mask  = out + 3 * BW;                         // [B,W]
    float* iw    = out + 4 * BW;                         // [B,W]
    float* scal  = out + 5 * BW;                         // [2] scalars

    int* hdr     = (int*)d_ws;                           // [0]=num_samples [1]=pos [2]=neg
    int* rowoff  = hdr + 16;                             // [B]  rowsum -> excl offsets
    int* chunk   = rowoff + B;                           // [CH] counts -> excl offsets
    (void)ws_size; (void)n_in; (void)out_size;

    // K0: zero mask + delta + inside_weights (4*BW floats) as float4
    long long n4 = BW;                                   // 4*BW/4
    k0_zero<<<(unsigned)((n4 + 255) / 256), 256, 0, stream>>>((float4*)(out + BW), n4);
    // K1: scatter + row positive counts
    k1_scatter<<<B, N, 0, stream>>>(slp, mask, delta, rowoff, N, W);
    // K2: row scan -> offsets, num_positive
    k2_scanrows<<<1, 1024, 0, stream>>>(rowoff, B, hdr, scal);
    // K3: cls goals + delta transform
    k3_elemwise<<<(unsigned)((BW + 255) / 256), 256, 0, stream>>>(cls, delta, mask, BW, W);
    // K4: negative chunk counts (permuted order)
    k4_negchunks<<<CH, 256, 0, stream>>>(mask, M, pmask, chunk);
    // K5: chunk scan -> offsets, num_negative, num_samples
    k5_scanchunks<<<1, 1024, 0, stream>>>(chunk, CH, hdr, scal);
    // K6: positive selection (WMMA prefix scan)
    k6_posselect<<<B, N, 0, stream>>>(slp, iw, rowoff, hdr, N, W);
    // K7: negative selection
    k7_negselect<<<CH, 256, 0, stream>>>(mask, iw, chunk, hdr, M, pmask);
}
